// PN_Critic_72997264163352
// MI455X (gfx1250) — compile-verified, matrix operand order checked by reading, not run on hardware
//
#include <hip/hip_runtime.h>
#include <hip/hip_bf16.h>
#include <math.h>

// Sizes from the reference
#define B_   256
#define S_   512
#define D_   128
#define H1_  256
#define H2_  512
#define NG   2048   // 4*H2

typedef __attribute__((ext_vector_type(16))) _Float16 v16h;
typedef __attribute__((ext_vector_type(8)))  float    v8f;

// ---------------- workspace layout (bytes, all 256B aligned) ----------------
#define OFF_WHH_FRAG    ((size_t)0)
#define OFF_WCOMB_FRAG  ((size_t)(2u*1024u*1024u))
#define OFF_BIAS        (OFF_WCOMB_FRAG + 524288u)
#define OFF_WCOMB_F32   (OFF_BIAS + 8192u)
#define OFF_POOLED      (OFF_WCOMB_F32 + 1048576u)

// ---------------------------------------------------------------------------
// Prep: W_comb[n][d] = sum_h w_ih[n][h] * conv_w[h][d]
__global__ __launch_bounds__(256) void wcomb_kernel(
    const float* __restrict__ w_ih, const float* __restrict__ conv_w,
    float* __restrict__ W_comb) {
  int idx = blockIdx.x * 256 + threadIdx.x;     // 2048*128
  int n = idx >> 7, d = idx & 127;
  float s = 0.f;
  for (int h = 0; h < H1_; ++h)
    s += w_ih[(size_t)n * H1_ + h] * conv_w[(size_t)h * D_ + d];
  W_comb[idx] = s;
}

// bias_comb[n] = b_ih[n] + b_hh[n] + sum_h w_ih[n][h]*conv_b[h]
__global__ __launch_bounds__(256) void bias_kernel(
    const float* __restrict__ w_ih, const float* __restrict__ conv_b,
    const float* __restrict__ b_ih, const float* __restrict__ b_hh,
    float* __restrict__ bias) {
  int n = blockIdx.x * 256 + threadIdx.x;       // 2048
  float s = b_ih[n] + b_hh[n];
  for (int h = 0; h < H1_; ++h)
    s += w_ih[(size_t)n * H1_ + h] * conv_b[h];
  bias[n] = s;
}

// Pack a row-major [NG][K] f32 matrix into WMMA B-fragment order (f16):
// fragment (kt,nt): lane l holds B col n = nt*16 + (l&15),
// K chunk kb = kt*32 + (l>>4)*16, 16 consecutive k values (contig in source row).
__global__ __launch_bounds__(256) void pack_bfrag_kernel(
    const float* __restrict__ src, _Float16* __restrict__ dst, int K) {
  int tid  = blockIdx.x * 256 + threadIdx.x;    // nfrag*32
  int lane = tid & 31, frag = tid >> 5;
  int kt = frag / 128, nt = frag % 128;
  int n  = nt * 16 + (lane & 15);
  int kb = kt * 32 + ((lane >> 4) * 16);
  const float* s = src + (size_t)n * K + kb;
  _Float16* d = dst + (size_t)frag * 512 + lane * 16;
#pragma unroll
  for (int e = 0; e < 16; ++e) d[e] = (_Float16)s[e];
}

// ---------------------------------------------------------------------------
// Main recurrent kernel: 16 blocks (one per 16-row batch tile) x 256 threads.
// 8 waves; wave w owns hidden slice j in [w*64, w*64+64) for ALL four gates.
__global__ __launch_bounds__(256) void lstm_kernel(
    const float*    __restrict__ x,           // [B,S,D] f32
    const _Float16* __restrict__ whh_frag,    // packed, K=512
    const _Float16* __restrict__ wcomb_frag,  // packed, K=128
    const float*    __restrict__ bias_comb,   // [2048]
    const float*    __restrict__ h0,          // [B,H2]
    const float*    __restrict__ c0,          // [B,H2]
    float*          __restrict__ pooled) {    // [B,H2]
  __shared__ __align__(32) _Float16 h_lds[16 * H2_];   // 16 KB
  __shared__ __align__(32) _Float16 x_lds[16 * D_];    // 4 KB

  const int tid   = threadIdx.x;
  const int lane  = tid & 31;
  const int w     = tid >> 5;          // wave 0..7
  const int bbase = blockIdx.x * 16;

  // C/D fragment element mapping: VGPR r, lane l -> M = r + 8*(l>>4), N = l&15
  const int brow = (lane >> 4) * 8;
  const int ncol = lane & 15;

  // Per-lane bias registers: acc k16 = g*4+jt covers n = g*512 + w*64 + jt*16 + ncol
  float biasr[16];
#pragma unroll
  for (int g = 0; g < 4; ++g)
#pragma unroll
    for (int jt = 0; jt < 4; ++jt)
      biasr[g * 4 + jt] = bias_comb[g * H2_ + w * 64 + jt * 16 + ncol];

  // Resident cell state + pooled sum (32 elements each, in VGPRs all 512 steps)
  float cst[32], pool[32];
#pragma unroll
  for (int jt = 0; jt < 4; ++jt)
#pragma unroll
    for (int r = 0; r < 8; ++r) {
      int b = brow + r;
      int j = w * 64 + jt * 16 + ncol;
      cst[jt * 8 + r]  = c0[(size_t)(bbase + b) * H2_ + j];
      pool[jt * 8 + r] = 0.f;
      h_lds[b * H2_ + j] = (_Float16)h0[(size_t)(bbase + b) * H2_ + j];
    }
  __syncthreads();

  // Per-wave fragment base pointers: this wave's 16 ntiles are
  // nt(k16 = g*4+jt) = g*32 + w*4 + jt  (512/16 = 32 ntiles per gate group)
  const _Float16* whh_lane = whh_frag   + (size_t)lane * 16 + (size_t)(w * 4) * 512;
  const _Float16* wc_lane  = wcomb_frag + (size_t)lane * 16 + (size_t)(w * 4) * 512;

  const int arow = lane & 15;
  const int asel = (lane >> 4) * 16;

#pragma unroll 1
  for (int t = 0; t < S_; ++t) {
    // Opaque zero, regenerated every iteration: defeats LICM on the B-fragment
    // loads (otherwise LLVM hoists all 320 loop-invariant loads and spills),
    // while keeping them plain cache-friendly global_load_b128 that the
    // scheduler can clause and overlap with WMMA (unlike volatile loads,
    // which serialized the stream with s_wait_loadcnt 0 between each pair).
    unsigned zoff;
    asm volatile("v_mov_b32 %0, 0" : "=v"(zoff));
    const _Float16* wc_it  = wc_lane  + zoff;
    const _Float16* whh_it = whh_lane + zoff;

    // Stage x_t tile [16][128] -> LDS f16 (each thread: 8 contiguous values)
    {
      int b  = tid >> 4;
      int d0 = (tid & 15) * 8;
      const float* src = x + ((size_t)(bbase + b) * S_ + t) * D_ + d0;
#pragma unroll
      for (int e = 0; e < 8; ++e) x_lds[b * D_ + d0 + e] = (_Float16)src[e];
      if (t + 1 < S_) __builtin_prefetch(src + D_, 0, 1);  // next step's x row
    }
    __syncthreads();  // x staged + previous step's h writes visible

    // acc init with fused bias
    v8f acc[16];
#pragma unroll
    for (int k16 = 0; k16 < 16; ++k16) {
      float bv = biasr[k16];
#pragma unroll
      for (int r = 0; r < 8; ++r) acc[k16][r] = bv;
    }

    // Input contribution: x_t @ W_comb.T  (K = 128, 4 ktiles)
#pragma unroll
    for (int kt = 0; kt < 4; ++kt) {
      v16h a = *(const v16h*)&x_lds[arow * D_ + kt * 32 + asel];
#pragma unroll
      for (int g = 0; g < 4; ++g)
#pragma unroll
        for (int jt = 0; jt < 4; ++jt) {
          v16h bf = *(const v16h*)(wc_it + (size_t)(kt * 128 + g * 32 + jt) * 512);
          acc[g * 4 + jt] = __builtin_amdgcn_wmma_f32_16x16x32_f16(
              false, a, false, bf, (short)0, acc[g * 4 + jt], false, false);
        }
    }
    // Recurrent contribution: h @ w_hh.T  (K = 512, 16 ktiles)
#pragma unroll
    for (int kt = 0; kt < 16; ++kt) {
      v16h a = *(const v16h*)&h_lds[arow * H2_ + kt * 32 + asel];
#pragma unroll
      for (int g = 0; g < 4; ++g)
#pragma unroll
        for (int jt = 0; jt < 4; ++jt) {
          v16h bf = *(const v16h*)(whh_it + (size_t)(kt * 128 + g * 32 + jt) * 512);
          acc[g * 4 + jt] = __builtin_amdgcn_wmma_f32_16x16x32_f16(
              false, a, false, bf, (short)0, acc[g * 4 + jt], false, false);
        }
    }

    __syncthreads();  // everyone done reading h_lds before it is overwritten

    // LSTM cell update, purely per-lane (gates i/f/g/o live in this wave's accs)
#pragma unroll
    for (int jt = 0; jt < 4; ++jt)
#pragma unroll
      for (int r = 0; r < 8; ++r) {
        float iv = acc[0 * 4 + jt][r];
        float fv = acc[1 * 4 + jt][r];
        float gv = acc[2 * 4 + jt][r];
        float ov = acc[3 * 4 + jt][r];
        iv = 1.f / (1.f + __expf(-iv));
        fv = 1.f / (1.f + __expf(-fv));
        gv = tanhf(gv);
        ov = 1.f / (1.f + __expf(-ov));
        float c = fv * cst[jt * 8 + r] + iv * gv;
        cst[jt * 8 + r] = c;
        float h = ov * tanhf(c);
        pool[jt * 8 + r] += h;
        int b = brow + r;
        int j = w * 64 + jt * 16 + ncol;
        h_lds[b * H2_ + j] = (_Float16)h;   // feeds next step's A fragments
      }
  }

  // Write pooled sums
#pragma unroll
  for (int jt = 0; jt < 4; ++jt)
#pragma unroll
    for (int r = 0; r < 8; ++r) {
      int b = brow + r;
      int j = w * 64 + jt * 16 + ncol;
      pooled[(size_t)(bbase + b) * H2_ + j] = pool[jt * 8 + r];
    }
}

// ---------------------------------------------------------------------------
// Head: v[b] = fc2_b + sum_j relu(pooled[b]@fc1_w[j] + fc1_b[j]) * fc2_w[j]
__global__ __launch_bounds__(256) void head_kernel(
    const float* __restrict__ pooled, const float* __restrict__ fc1_w,
    const float* __restrict__ fc1_b, const float* __restrict__ fc2_w,
    const float* __restrict__ fc2_b, float* __restrict__ out) {
  __shared__ float row[H2_];
  __shared__ float red[256];
  int b = blockIdx.x, tid = threadIdx.x;
  row[tid]       = pooled[(size_t)b * H2_ + tid];
  row[tid + 256] = pooled[(size_t)b * H2_ + tid + 256];
  __syncthreads();
  float acc = 0.f;
#pragma unroll
  for (int jj = 0; jj < 2; ++jj) {
    int j = tid + jj * 256;
    const float* wr = fc1_w + (size_t)j * H2_;
    float s = fc1_b[j];
    for (int k = 0; k < H2_; ++k) s += row[k] * wr[k];
    s = fmaxf(s, 0.f);
    acc += s * fc2_w[j];
  }
  red[tid] = acc;
  __syncthreads();
  for (int off = 128; off > 0; off >>= 1) {
    if (tid < off) red[tid] += red[tid + off];
    __syncthreads();
  }
  if (tid == 0) out[b] = red[0] + fc2_b[0];
}

// ---------------------------------------------------------------------------
extern "C" void kernel_launch(void* const* d_in, const int* in_sizes, int n_in,
                              void* d_out, int out_size, void* d_ws, size_t ws_size,
                              hipStream_t stream) {
  (void)in_sizes; (void)n_in; (void)out_size; (void)ws_size;
  const float* x      = (const float*)d_in[0];
  const float* conv_w = (const float*)d_in[1];
  const float* conv_b = (const float*)d_in[2];
  const float* w_ih   = (const float*)d_in[3];
  const float* b_ih   = (const float*)d_in[4];
  const float* w_hh   = (const float*)d_in[5];
  const float* b_hh   = (const float*)d_in[6];
  const float* h0     = (const float*)d_in[7];
  const float* c0     = (const float*)d_in[8];
  const float* fc1_w  = (const float*)d_in[9];
  const float* fc1_b  = (const float*)d_in[10];
  const float* fc2_w  = (const float*)d_in[11];
  const float* fc2_b  = (const float*)d_in[12];
  float* out = (float*)d_out;

  char* ws = (char*)d_ws;
  _Float16* whh_frag   = (_Float16*)(ws + OFF_WHH_FRAG);
  _Float16* wcomb_frag = (_Float16*)(ws + OFF_WCOMB_FRAG);
  float*    bias_comb  = (float*)(ws + OFF_BIAS);
  float*    W_comb     = (float*)(ws + OFF_WCOMB_F32);
  float*    pooled     = (float*)(ws + OFF_POOLED);

  // Prep (stream-ordered dependencies)
  wcomb_kernel<<<(NG * D_) / 256, 256, 0, stream>>>(w_ih, conv_w, W_comb);
  bias_kernel<<<NG / 256, 256, 0, stream>>>(w_ih, conv_b, b_ih, b_hh, bias_comb);
  // w_hh: 16 ktiles * 128 ntiles fragments
  pack_bfrag_kernel<<<(16 * 128 * 32) / 256, 256, 0, stream>>>(w_hh, whh_frag, H2_);
  // W_comb: 4 ktiles * 128 ntiles fragments
  pack_bfrag_kernel<<<(4 * 128 * 32) / 256, 256, 0, stream>>>(W_comb, wcomb_frag, D_);

  // Recurrent core: 16 independent batch tiles
  lstm_kernel<<<16, 256, 0, stream>>>(x, whh_frag, wcomb_frag, bias_comb, h0, c0, pooled);

  // Head
  head_kernel<<<B_, 256, 0, stream>>>(pooled, fc1_w, fc1_b, fc2_w, fc2_b, out);
}
